// EncodecEuclideanCodebook_30339648979662
// MI455X (gfx1250) — matrix operand order repaired and checked by reading, not compiled, not run on hardware
//
#include <hip/hip_runtime.h>
#include <hip/hip_bf16.h>

typedef __attribute__((ext_vector_type(16))) __bf16 v16bf;
typedef __attribute__((ext_vector_type(8)))  float  v8f;
typedef __attribute__((ext_vector_type(4)))  float  v4f;

#define DIM   128
#define TILE_M 16
#define TILE_N 16
#define WAVES_PER_BLOCK 8

// ---------------------------------------------------------------------------
// Kernel 1: split embed (f32) into bf16 hi/lo planes + f32 row norms.
// 1024 blocks x 32 lanes; each block handles one codebook row of 128 floats.
// ---------------------------------------------------------------------------
__global__ __launch_bounds__(32)
void vq_prep_kernel(const float* __restrict__ embed,
                    __bf16* __restrict__ ehi,
                    __bf16* __restrict__ elo,
                    float* __restrict__ esq) {
    const int bin  = blockIdx.x;
    const int lane = threadIdx.x;
    const float* row = embed + (size_t)bin * DIM;
    float ssum = 0.f;
#pragma unroll
    for (int j = 0; j < DIM / 32; ++j) {
        const int k = lane + j * 32;
        float v = row[k];
        __bf16 h = (__bf16)v;
        float  r = v - (float)h;
        ehi[(size_t)bin * DIM + k] = h;
        elo[(size_t)bin * DIM + k] = (__bf16)r;
        ssum += v * v;
    }
#pragma unroll
    for (int off = 1; off < 32; off <<= 1)
        ssum += __shfl_xor(ssum, off, 32);
    if (lane == 0) esq[bin] = ssum;
}

// ---------------------------------------------------------------------------
// Kernel 2: fused GEMM (x . E^T via split-bf16 WMMA) + argmax + gather.
// One wave per 16-row tile of x. 64 bin-tiles of 16, K=128 in 4 chunks of 32.
// ---------------------------------------------------------------------------
__global__ __launch_bounds__(WAVES_PER_BLOCK * 32)
void vq_main_kernel(const float* __restrict__ x,
                    const float* __restrict__ embed,
                    const __bf16* __restrict__ ehi,
                    const __bf16* __restrict__ elo,
                    const float* __restrict__ esq,
                    float* __restrict__ outq,
                    float* __restrict__ outind,
                    int nbins) {
    const int lane = threadIdx.x & 31;
    const int wave = threadIdx.x >> 5;
    const int tile = blockIdx.x * WAVES_PER_BLOCK + wave;
    const int row0 = tile * TILE_M;
    const int half = lane >> 4;   // which 16-lane half of the wave
    const int nloc = lane & 15;

    // ---- Build A fragments (16x32 bf16 per chunk), hi + lo split ----------
    // lanes 0-15 hold row M=lane, K in {0..15} (+32*c);
    // lanes 16-31 hold row M=lane-16, K in {16..31} (+32*c).
    v16bf ahi[4], alo[4];
    const float* xrow = x + (size_t)(row0 + nloc) * DIM;
#pragma unroll
    for (int c = 0; c < 4; ++c) {
        const int kbase = c * 32 + half * 16;
        v4f f0 = __builtin_nontemporal_load(reinterpret_cast<const v4f*>(xrow + kbase));
        v4f f1 = __builtin_nontemporal_load(reinterpret_cast<const v4f*>(xrow + kbase + 4));
        v4f f2 = __builtin_nontemporal_load(reinterpret_cast<const v4f*>(xrow + kbase + 8));
        v4f f3 = __builtin_nontemporal_load(reinterpret_cast<const v4f*>(xrow + kbase + 12));
        float fv[16] = { f0[0], f0[1], f0[2], f0[3],  f1[0], f1[1], f1[2], f1[3],
                         f2[0], f2[1], f2[2], f2[3],  f3[0], f3[1], f3[2], f3[3] };
        v16bf h, l;
#pragma unroll
        for (int e = 0; e < 16; ++e) {
            __bf16 hb = (__bf16)fv[e];
            h[e] = hb;
            l[e] = (__bf16)(fv[e] - (float)hb);
        }
        ahi[c] = h;
        alo[c] = l;
    }

    // Running per-lane argmax. VGPR r of the accumulator holds row
    // M = r + 8*half, column n = nloc of the current bin tile.
    float bestS[8];
    int   bestI[8];
#pragma unroll
    for (int r = 0; r < 8; ++r) { bestS[r] = -3.0e38f; bestI[r] = 0; }

    const int ntiles = nbins / TILE_N;
    for (int t = 0; t < ntiles; ++t) {
        // B fragment (32x16): lane holds column n=nloc, K = half*16 + e (+32*c)
        // -> 16 consecutive bf16 = one 32B contiguous load per lane.
        const __bf16* bh = ehi + ((size_t)(t * TILE_N + nloc)) * DIM + half * 16;
        const __bf16* bl = elo + ((size_t)(t * TILE_N + nloc)) * DIM + half * 16;

        v8f acc = {};
#pragma unroll
        for (int c = 0; c < 4; ++c) {
            v16bf bhv = *reinterpret_cast<const v16bf*>(bh + c * 32);
            v16bf blv = *reinterpret_cast<const v16bf*>(bl + c * 32);
            // split-precision product: xh*eh + xh*el + xl*eh  (~f32 accuracy)
            acc = __builtin_amdgcn_wmma_f32_16x16x32_bf16(
                false, ahi[c], false, bhv, (short)0, acc, false, false);
            acc = __builtin_amdgcn_wmma_f32_16x16x32_bf16(
                false, ahi[c], false, blv, (short)0, acc, false, false);
            acc = __builtin_amdgcn_wmma_f32_16x16x32_bf16(
                false, alo[c], false, bhv, (short)0, acc, false, false);
        }

        const float en   = esq[t * TILE_N + nloc];
        const int   cand = t * TILE_N + nloc;
#pragma unroll
        for (int r = 0; r < 8; ++r) {
            float s = 2.0f * acc[r] - en;          // = -(||x-e||^2) + ||x||^2
            if (s > bestS[r]) { bestS[r] = s; bestI[r] = cand; }
        }
    }

    // ---- Cross-lane argmax within each 16-lane half (rows r / r+8) --------
#pragma unroll
    for (int r = 0; r < 8; ++r) {
        float s = bestS[r];
        int   i = bestI[r];
#pragma unroll
        for (int off = 1; off < 16; off <<= 1) {
            float so = __shfl_xor(s, off, 32);
            int   io = __shfl_xor(i, off, 32);
            if (so > s || (so == s && io < i)) { s = so; i = io; }
        }
        bestS[r] = s;
        bestI[r] = i;
    }

    // ---- Write indices (as float, matching d_out dtype) -------------------
    if (nloc == 0) {
#pragma unroll
        for (int r = 0; r < 8; ++r)
            outind[row0 + half * 8 + r] = (float)bestI[r];
    }

    // ---- Gather winning codebook rows: all 32 lanes copy 128 f32 per row --
#pragma unroll
    for (int r = 0; r < 8; ++r) {
        const int i0 = __shfl(bestI[r], 0, 32);    // row r      (half 0)
        const int i1 = __shfl(bestI[r], 16, 32);   // row r + 8  (half 1)
        v4f q0 = *reinterpret_cast<const v4f*>(embed + (size_t)i0 * DIM + lane * 4);
        __builtin_nontemporal_store(
            q0, reinterpret_cast<v4f*>(outq + (size_t)(row0 + r) * DIM + lane * 4));
        v4f q1 = *reinterpret_cast<const v4f*>(embed + (size_t)i1 * DIM + lane * 4);
        __builtin_nontemporal_store(
            q1, reinterpret_cast<v4f*>(outq + (size_t)(row0 + 8 + r) * DIM + lane * 4));
    }
}

// ---------------------------------------------------------------------------
extern "C" void kernel_launch(void* const* d_in, const int* in_sizes, int n_in,
                              void* d_out, int out_size, void* d_ws, size_t ws_size,
                              hipStream_t stream) {
    const float* x     = (const float*)d_in[0];
    const float* embed = (const float*)d_in[1];

    const int N     = in_sizes[0] / DIM;   // 131072 rows
    const int BINS  = in_sizes[1] / DIM;   // 1024 codes

    // workspace layout: ehi | elo | esq
    char* ws = (char*)d_ws;
    __bf16* ehi = (__bf16*)ws;                                   // BINS*DIM*2 B
    __bf16* elo = (__bf16*)(ws + (size_t)BINS * DIM * 2);        // BINS*DIM*2 B
    float*  esq = (float*)(ws + (size_t)BINS * DIM * 4);         // BINS*4 B

    float* outq   = (float*)d_out;                 // [N, DIM] quantize
    float* outind = outq + (size_t)N * DIM;        // [N] indices (as float)

    vq_prep_kernel<<<BINS, 32, 0, stream>>>(embed, ehi, elo, esq);

    const int tiles  = N / TILE_M;                 // 8192
    const int blocks = tiles / WAVES_PER_BLOCK;    // 1024
    vq_main_kernel<<<blocks, WAVES_PER_BLOCK * 32, 0, stream>>>(
        x, embed, ehi, elo, esq, outq, outind, BINS);
}
